// SelfAttention_14697378087570
// MI455X (gfx1250) — matrix operand order repaired and checked by reading, not compiled, not run on hardware
//
#include <hip/hip_runtime.h>

#define NUM_HEADS 12
#define HIDDEN    768
#define HEAD_DIM  64
#define BATCH     2
#define SEQ       2048
#define ROWS      (BATCH*SEQ)   // 4096

typedef __bf16 bf16_t;
typedef bf16_t v16bf __attribute__((ext_vector_type(16)));
typedef bf16_t v8bf  __attribute__((ext_vector_type(8)));
typedef bf16_t v4bf  __attribute__((ext_vector_type(4)));
typedef float  v8f   __attribute__((ext_vector_type(8)));
typedef float  v4f   __attribute__((ext_vector_type(4)));

static __device__ __forceinline__ v8f wmma_bf16(v16bf a, v16bf b, v8f c) {
    return __builtin_amdgcn_wmma_f32_16x16x32_bf16(false, a, false, b, (short)0, c, false, false);
}

// Async global->LDS 16-byte copy (CDNA5 GLOBAL_LOAD_ASYNC_TO_LDS_B128, GV mode).
// lds_off: byte offset of destination in LDS (low 32 bits of generic pointer);
// gaddr: 64-bit global address. Tracked by ASYNCcnt.
static __device__ __forceinline__ void async_copy_b128(unsigned lds_off, const void* gaddr) {
    asm volatile("global_load_async_to_lds_b128 %0, %1, off"
                 :: "v"(lds_off), "v"(gaddr) : "memory");
}
static __device__ __forceinline__ void wait_asynccnt0() {
    asm volatile("s_wait_asynccnt 0x0" ::: "memory");
}

// ---------------------------------------------------------------------------
// Prep kernel A: elementwise f32 -> bf16 (hidden_states -> Xbf).
// ---------------------------------------------------------------------------
__global__ __launch_bounds__(256)
void cvt_bf16(const float* __restrict__ src, bf16_t* __restrict__ dst, int n4) {
    int i = blockIdx.x * 256 + threadIdx.x;
    if (i < n4) {
        v4f f = ((const v4f*)src)[i];
        v4bf t;
        t[0] = (bf16_t)f[0]; t[1] = (bf16_t)f[1];
        t[2] = (bf16_t)f[2]; t[3] = (bf16_t)f[3];
        ((v4bf*)dst)[i] = t;
    }
}

// ---------------------------------------------------------------------------
// Prep kernel B: transpose + convert weights.  Wt3[z][n][k] = bf16(W_z[k][n]).
// 32x32 LDS tile transpose; coalesced reads and writes.
// ---------------------------------------------------------------------------
__global__ __launch_bounds__(256)
void wt_convert(const float* __restrict__ Wq, const float* __restrict__ Wk,
                const float* __restrict__ Wv, bf16_t* __restrict__ Wt3) {
    __shared__ float tile[32][33];   // +1 pad: no bank conflicts
    const int z = blockIdx.z;
    const float* W = (z == 0) ? Wq : (z == 1) ? Wk : Wv;
    bf16_t* Out = Wt3 + (size_t)z * HIDDEN * HIDDEN;
    const int k0 = blockIdx.x * 32, n0 = blockIdx.y * 32;
    const int c = threadIdx.x & 31, r4 = (threadIdx.x >> 5) * 4;
    #pragma unroll
    for (int i = 0; i < 4; ++i)
        tile[r4 + i][c] = W[(size_t)(k0 + r4 + i) * HIDDEN + n0 + c];
    __syncthreads();
    #pragma unroll
    for (int i = 0; i < 4; ++i)
        Out[(size_t)(n0 + r4 + i) * HIDDEN + k0 + c] = (bf16_t)tile[c][r4 + i];
}

// ---------------------------------------------------------------------------
// Kernel 1: fused QKV projection.  Y = X @ W + b for W in {Wq,Wk,Wv}.
// X: [4096,768] bf16; W^T: [768 n][768 k] bf16 -> both tiles staged with
// async-to-LDS b128 copies (no conversion in the hot loop).
// Q/K stored bf16 [B,H,S,d]; V stored bf16 [B,H,d,S] (d-major for attention).
// Block tile 128(M) x 128(N), K-step 32; 8 waves as 4(M) x 2(N); each wave
// 2x4 WMMA tiles = 8 WMMAs per K-step.
// ---------------------------------------------------------------------------
__global__ __launch_bounds__(256)
void qkv_gemm(const bf16_t* __restrict__ X,
              const bf16_t* __restrict__ Wt3,
              const float* __restrict__ bq, const float* __restrict__ bk,
              const float* __restrict__ bv,
              bf16_t* __restrict__ Qw, bf16_t* __restrict__ Kw,
              bf16_t* __restrict__ Vw)
{
    __shared__ __align__(16) bf16_t Xs[128 * 32];   // [row][k] row-major, 8 KB
    __shared__ __align__(16) bf16_t Wt[128 * 32];   // [n][k]   transposed, 8 KB

    const int z = blockIdx.z;
    const bf16_t* Wg   = Wt3 + (size_t)z * HIDDEN * HIDDEN;   // [n][k]
    const float*  bias = (z == 0) ? bq : (z == 1) ? bk : bv;
    bf16_t*       Out  = (z == 0) ? Qw : (z == 1) ? Kw : Vw;
    const bool vtrans  = (z == 2);

    const int m0   = blockIdx.x * 128;
    const int n0   = blockIdx.y * 128;
    const int tid  = threadIdx.x;
    const int lane = tid & 31;
    const int w    = tid >> 5;       // 0..7
    const int wm   = (w & 3) * 32;   // wave M offset in block tile
    const int wn   = (w >> 2) * 64;  // wave N offset in block tile

    v8f acc[2][4] = {};

    const int akb = (lane >= 16) ? 8  : 0;   // A-frag K base (16-bit layout)
    const int bkb = (lane >= 16) ? 8  : 0;   // B-frag K base

    for (int k0 = 0; k0 < HIDDEN; k0 += 32) {
        __syncthreads();
        // ---- async stage X tile [128][32] and W^T tile [128][32] (bf16)
        #pragma unroll
        for (int it = 0; it < 2; ++it) {
            int id = tid + it * 256;          // 0..511
            int r  = id >> 2;                 // 0..127
            int c  = (id & 3) * 8;            // 0,8,16,24
            async_copy_b128((unsigned)(size_t)&Xs[r * 32 + c],
                            &X[(size_t)(m0 + r) * HIDDEN + k0 + c]);
            async_copy_b128((unsigned)(size_t)&Wt[r * 32 + c],
                            &Wg[(size_t)(n0 + r) * HIDDEN + k0 + c]);
        }
        // ---- prefetch next K-step tiles into cache (global_prefetch_b8)
        if (k0 + 32 < HIDDEN) {
            int r = tid & 127;
            __builtin_prefetch(&X[(size_t)(m0 + r) * HIDDEN + k0 + 32], 0, 3);
            __builtin_prefetch(&Wg[(size_t)(n0 + r) * HIDDEN + k0 + 32], 0, 3);
        }
        wait_asynccnt0();
        __syncthreads();

        // ---- gather fragments and multiply
        v16bf a[2], b[4];
        #pragma unroll
        for (int mf = 0; mf < 2; ++mf) {
            int r = wm + mf * 16 + (lane & 15);
            union { v8bf h[2]; v16bf v; } u;
            u.h[0] = *(const v8bf*)&Xs[r * 32 + akb];
            u.h[1] = *(const v8bf*)&Xs[r * 32 + akb + 16];
            a[mf] = u.v;
        }
        #pragma unroll
        for (int nf = 0; nf < 4; ++nf) {
            int n = wn + nf * 16 + (lane & 15);
            union { v8bf h[2]; v16bf v; } u;
            u.h[0] = *(const v8bf*)&Wt[n * 32 + bkb];
            u.h[1] = *(const v8bf*)&Wt[n * 32 + bkb + 16];
            b[nf] = u.v;
        }
        #pragma unroll
        for (int mf = 0; mf < 2; ++mf)
            #pragma unroll
            for (int nf = 0; nf < 4; ++nf)
                acc[mf][nf] = wmma_bf16(a[mf], b[nf], acc[mf][nf]);
    }

    // ---- epilogue: bias add, convert, store
    const int rl = (lane >> 4) << 3;   // +8 for upper half lanes
    #pragma unroll
    for (int mf = 0; mf < 2; ++mf)
        #pragma unroll
        for (int nf = 0; nf < 4; ++nf)
            #pragma unroll
            for (int i = 0; i < 8; ++i) {
                int gm = m0 + wm + mf * 16 + i + rl;
                int gn = n0 + wn + nf * 16 + (lane & 15);
                float vv = acc[mf][nf][i] + bias[gn];
                int bb = gm >> 11, s = gm & 2047;
                int hh = gn >> 6,  dd = gn & 63;
                size_t idx = vtrans
                    ? ((size_t)(bb * NUM_HEADS + hh) * HEAD_DIM + dd) * SEQ + s
                    : ((size_t)(bb * NUM_HEADS + hh) * SEQ + s) * HEAD_DIM + dd;
                Out[idx] = (bf16_t)vv;
            }
}

// ---------------------------------------------------------------------------
// Kernel 2: flash attention.  One block per (b,h, 128-query tile).
// 8 waves x 16 query rows.  Per 32-key block: async-copy K [32][64] and
// V^T [64][32] tiles into LDS, 4 score WMMAs + online softmax + 4 PV WMMAs.
// ---------------------------------------------------------------------------
__global__ __launch_bounds__(256)
void flash_attn(const bf16_t* __restrict__ Q, const bf16_t* __restrict__ K,
                const bf16_t* __restrict__ V, const float* __restrict__ mask,
                float* __restrict__ out)
{
    __shared__ __align__(16) bf16_t Kb[32 * 64];     // [s][d] row-major, 4 KB
    __shared__ __align__(16) bf16_t Vt[64 * 32];     // [d][s], 4 KB
    __shared__ __align__(16) bf16_t Ps[8][16 * 32];  // per-wave P scratch, 8 KB

    const int bh   = blockIdx.y;                 // 0..23
    const int b    = bh / NUM_HEADS;
    const int h    = bh % NUM_HEADS;
    const int tid  = threadIdx.x;
    const int lane = tid & 31;
    const int w    = tid >> 5;
    const int q0   = blockIdx.x * 128 + w * 16;  // this wave's query rows

    const bf16_t* Qh = Q + (size_t)bh * SEQ * HEAD_DIM;   // [S][d]
    const bf16_t* Kh = K + (size_t)bh * SEQ * HEAD_DIM;   // [S][d]
    const bf16_t* Vh = V + (size_t)bh * HEAD_DIM * SEQ;   // [d][S] (d-major)

    const int akb = (lane >= 16) ? 8  : 0;
    const int bkb = (lane >= 16) ? 8  : 0;

    // ---- load Q fragments once (A-layout, two K=32 chunks over d=64)
    v16bf qa[2];
    #pragma unroll
    for (int f = 0; f < 2; ++f) {
        const bf16_t* qrow = Qh + (size_t)(q0 + (lane & 15)) * HEAD_DIM + f * 32 + akb;
        union { v8bf h2[2]; v16bf v; } u;
        u.h2[0] = *(const v8bf*)qrow;
        u.h2[1] = *(const v8bf*)(qrow + 16);
        qa[f] = u.v;
    }

    v8f acc[4] = {};
    float mrun[8], lrun[8];
    #pragma unroll
    for (int i = 0; i < 8; ++i) { mrun[i] = -1.0e30f; lrun[i] = 0.0f; }

    const float scale = 0.125f;   // 1/sqrt(64)

    // per-thread async-copy coordinates (256 threads x 16B = one 4 KB tile)
    const int kr = tid >> 3, kc = (tid & 7) * 8;   // K tile [32][64]
    const int vd = tid >> 2, vc = (tid & 3) * 8;   // V tile [64][32]

    for (int kb = 0; kb < SEQ; kb += 32) {
        __syncthreads();
        // ---- async stage K block [32][64] and V^T block [64][32]
        async_copy_b128((unsigned)(size_t)&Kb[kr * 64 + kc],
                        Kh + (size_t)(kb + kr) * HEAD_DIM + kc);
        async_copy_b128((unsigned)(size_t)&Vt[vd * 32 + vc],
                        Vh + (size_t)vd * SEQ + kb + vc);
        wait_asynccnt0();
        __syncthreads();

        // ---- scores: two 16x16 tiles over this 32-key block
        v8f sc[2];
        #pragma unroll
        for (int nt = 0; nt < 2; ++nt) {
            v8f c = {};
            #pragma unroll
            for (int f = 0; f < 2; ++f) {
                int n = nt * 16 + (lane & 15);   // key row in block
                union { v8bf h2[2]; v16bf v; } u;
                u.h2[0] = *(const v8bf*)&Kb[n * 64 + f * 32 + bkb];
                u.h2[1] = *(const v8bf*)&Kb[n * 64 + f * 32 + bkb + 16];
                c = wmma_bf16(qa[f], u.v, c);
            }
            float mval = mask[b * SEQ + kb + nt * 16 + (lane & 15)];
            #pragma unroll
            for (int i = 0; i < 8; ++i) c[i] = c[i] * scale + mval;
            sc[nt] = c;
        }

        // ---- online softmax (rows live across 16-lane N groups)
        #pragma unroll
        for (int i = 0; i < 8; ++i) {
            float pm = fmaxf(sc[0][i], sc[1][i]);
            #pragma unroll
            for (int m = 1; m <= 8; m <<= 1) pm = fmaxf(pm, __shfl_xor(pm, m, 32));
            float mnew = fmaxf(mrun[i], pm);
            float corr = __expf(mrun[i] - mnew);
            float p0 = __expf(sc[0][i] - mnew);
            float p1 = __expf(sc[1][i] - mnew);
            float rs = p0 + p1;
            #pragma unroll
            for (int m = 1; m <= 8; m <<= 1) rs += __shfl_xor(rs, m, 32);
            lrun[i] = lrun[i] * corr + rs;
            mrun[i] = mnew;
            #pragma unroll
            for (int nf = 0; nf < 4; ++nf) acc[nf][i] *= corr;
            // spill P (C layout) to per-wave LDS scratch as bf16
            int r = i + ((lane >> 4) << 3);
            Ps[w][r * 32 + (lane & 15)]      = (bf16_t)p0;
            Ps[w][r * 32 + 16 + (lane & 15)] = (bf16_t)p1;
        }

        // ---- reload P in A-fragment layout (16x32, K=32)
        union { v8bf h2[2]; v16bf v; } up;
        const bf16_t* prow = &Ps[w][(lane & 15) * 32 + akb];
        up.h2[0] = *(const v8bf*)prow;
        up.h2[1] = *(const v8bf*)(prow + 16);
        v16bf pa = up.v;

        // ---- O += P @ V  (4 column tiles across d=64)
        #pragma unroll
        for (int nf = 0; nf < 4; ++nf) {
            int n = nf * 16 + (lane & 15);   // d coordinate
            union { v8bf h2[2]; v16bf v; } u;
            u.h2[0] = *(const v8bf*)&Vt[n * 32 + bkb];
            u.h2[1] = *(const v8bf*)&Vt[n * 32 + bkb + 16];
            acc[nf] = wmma_bf16(pa, u.v, acc[nf]);
        }
    }

    // ---- finalize and store f32 ctx [B, S, HIDDEN]
    #pragma unroll
    for (int i = 0; i < 8; ++i) {
        int r = i + ((lane >> 4) << 3);
        int s = q0 + r;
        float inv = 1.0f / lrun[i];
        size_t base = ((size_t)b * SEQ + s) * HIDDEN + h * HEAD_DIM;
        #pragma unroll
        for (int nf = 0; nf < 4; ++nf)
            out[base + nf * 16 + (lane & 15)] = acc[nf][i] * inv;
    }
}

// ---------------------------------------------------------------------------
extern "C" void kernel_launch(void* const* d_in, const int* in_sizes, int n_in,
                              void* d_out, int out_size, void* d_ws, size_t ws_size,
                              hipStream_t stream) {
    const float* hidden = (const float*)d_in[0];
    const float* mask   = (const float*)d_in[1];
    const float* Wq     = (const float*)d_in[2];
    const float* bq     = (const float*)d_in[3];
    const float* Wk     = (const float*)d_in[4];
    const float* bk     = (const float*)d_in[5];
    const float* Wv     = (const float*)d_in[6];
    const float* bv     = (const float*)d_in[7];
    float* out = (float*)d_out;

    // workspace layout (bf16):
    //   Xbf  [4096][768]            6.29 MB
    //   Wt3  [3][768 n][768 k]      3.54 MB
    //   Qw   [B,H,S,d]              6.29 MB
    //   Kw   [B,H,S,d]              6.29 MB
    //   Vw   [B,H,d,S]              6.29 MB   (d-major)
    bf16_t* Xbf = (bf16_t*)d_ws;
    bf16_t* Wt3 = Xbf + (size_t)ROWS * HIDDEN;
    bf16_t* Qw  = Wt3 + (size_t)3 * HIDDEN * HIDDEN;
    bf16_t* Kw  = Qw + (size_t)ROWS * HIDDEN;
    bf16_t* Vw  = Kw + (size_t)ROWS * HIDDEN;

    int n4 = ROWS * HIDDEN / 4;
    cvt_bf16<<<(n4 + 255) / 256, 256, 0, stream>>>(hidden, Xbf, n4);

    dim3 gt(HIDDEN / 32, HIDDEN / 32, 3);     // 24 x 24 x 3
    wt_convert<<<gt, 256, 0, stream>>>(Wq, Wk, Wv, Wt3);

    dim3 gg(ROWS / 128, HIDDEN / 128, 3);     // 32 x 6 x 3
    qkv_gemm<<<gg, 256, 0, stream>>>(Xbf, Wt3, bq, bk, bv, Qw, Kw, Vw);

    dim3 ga(SEQ / 128, BATCH * NUM_HEADS);    // 16 x 24
    flash_attn<<<ga, 256, 0, stream>>>(Qw, Kw, Vw, mask, out);
}
